// GatedAttentionRNN_90477781057868
// MI455X (gfx1250) — matrix operand order repaired and checked by reading, not compiled, not run on hardware
//
#include <hip/hip_runtime.h>
#include <hip/hip_bf16.h>

// GatedAttentionRNN for MI455X (gfx1250, wave32).
// GEMMs: v_wmma_f32_16x16x32_bf16, fragments loaded directly from global with
// b128 loads (no LDS staging) -- per-lane fragment chunks are contiguous for
// row-major A[M,K] and W[N,K]. One wave computes a 32x32 tile (4 accumulators).
// GRU recurrence: Whh pinned in the 320KB WGP LDS (198KB, padded stride 129).

#define BB 8
#define LL 300
#define DD 256
#define HH 128
#define D2 512     // 2*D
#define H3 384     // 3*H
#define BL (BB*LL)
#define LQP 320    // padded Lq (rows of a)
#define LKP 320    // padded Lv (cols of a / K of context GEMM)
#define NEGF (-1e30f)

typedef __bf16 bf16;
typedef __attribute__((ext_vector_type(16))) __bf16 v16bf;
typedef __attribute__((ext_vector_type(8)))  __bf16 v8bf;
typedef __attribute__((ext_vector_type(8)))  float  v8f;

__device__ __forceinline__ float sigmoidf_(float x){ return 1.f/(1.f + __expf(-x)); }

// A fragment: lane (mr,half) holds row mr, K chunks [half*8,+8) and [16+half*8,+8)
__device__ __forceinline__ v16bf ld_a_frag(const bf16* __restrict__ A, int lda,
                                           int row, int k0, int half){
  const bf16* p = A + (long)row*lda + k0 + half*8;
  v8bf lo = *(const v8bf*)p;
  v8bf hi = *(const v8bf*)(p + 16);
  return __builtin_shufflevector(lo, hi, 0,1,2,3,4,5,6,7,8,9,10,11,12,13,14,15);
}
// B fragment (B=W^T, W row-major [N,K]): lane (mr,half) holds W[n0+mr], K chunks [half*8,+8) and [16+half*8,+8)
__device__ __forceinline__ v16bf ld_b_frag(const bf16* __restrict__ W, int ldb,
                                           int row, int k0, int half){
  const bf16* p = W + (long)row*ldb + k0 + half*8;
  v8bf lo = *(const v8bf*)p;
  v8bf hi = *(const v8bf*)(p + 16);
  return __builtin_shufflevector(lo, hi, 0,1,2,3,4,5,6,7,8,9,10,11,12,13,14,15);
}

// ---------------- elementwise pack f32 -> bf16 ----------------
__global__ void pack_bf16_k(const float* __restrict__ src, bf16* __restrict__ dst, int n){
  int i = blockIdx.x*blockDim.x + threadIdx.x;
  if (i < n) dst[i] = (bf16)src[i];
}

// rnn_in[:, 0:256] = u_query (context GEMM fills 256:512)
__global__ void concat_query_k(const float* __restrict__ uq, float* __restrict__ rnn){
  int i = blockIdx.x*blockDim.x + threadIdx.x;
  if (i < BL*DD){
    int row = i / DD, d = i % DD;
    rnn[(long)row*D2 + d] = uq[i];
  }
}

// uvT[b][d][v] = u_value[b][v][d] (zero-padded to LKP columns)
__global__ void transpose_uv_k(const bf16* __restrict__ uvb, bf16* __restrict__ uvT){
  int i = blockIdx.x*blockDim.x + threadIdx.x;
  if (i >= BB*DD*LKP) return;
  int v = i % LKP; int rest = i / LKP; int d = rest % DD; int b = rest / DD;
  bf16 val = (bf16)0.f;
  if (v < LL) val = uvb[((long)b*LL + v)*DD + d];
  uvT[i] = val;
}

// ---------------- WMMA GEMM: C[M,N] = A[M,K] @ W[N,K]^T (+bias) -------------
// Requirements (guaranteed by caller via buffer padding):
//   K % 32 == 0; lda, ldb % 8 == 0 (16B rows); all A/W tile reads in-bounds
//   for m0..m0+31 / n0..n0+31. Stores bounds-checked against M,N.
// One wave (32 threads) per 32x32 tile; batched via blockIdx.z.
__global__ void gemm_wmma_k(const bf16* __restrict__ A, const bf16* __restrict__ W,
                            const float* __restrict__ bias, float* __restrict__ C,
                            int M, int N, int K, int lda, int ldb, int ldc,
                            long sA, long sB, long sC){
  A += (long)blockIdx.z * sA;
  W += (long)blockIdx.z * sB;
  C += (long)blockIdx.z * sC;
  const int lane = threadIdx.x;           // 0..31
  const int mr = lane & 15, half = lane >> 4;
  const int m0 = blockIdx.x * 32, n0 = blockIdx.y * 32;
  v8f acc00 = {}, acc01 = {}, acc10 = {}, acc11 = {};
  for (int k0 = 0; k0 < K; k0 += 32){
    v16bf a0 = ld_a_frag(A, lda, m0 + mr,      k0, half);
    v16bf a1 = ld_a_frag(A, lda, m0 + 16 + mr, k0, half);
    v16bf b0 = ld_b_frag(W, ldb, n0 + mr,      k0, half);
    v16bf b1 = ld_b_frag(W, ldb, n0 + 16 + mr, k0, half);
    acc00 = __builtin_amdgcn_wmma_f32_16x16x32_bf16(false, a0, false, b0, (short)0, acc00, false, false);
    acc01 = __builtin_amdgcn_wmma_f32_16x16x32_bf16(false, a0, false, b1, (short)0, acc01, false, false);
    acc10 = __builtin_amdgcn_wmma_f32_16x16x32_bf16(false, a1, false, b0, (short)0, acc10, false, false);
    acc11 = __builtin_amdgcn_wmma_f32_16x16x32_bf16(false, a1, false, b1, (short)0, acc11, false, false);
  }
  // C/D layout: VGPR r -> row r + 8*half, col = lane&15 (within 16x16 tile)
  const int cn = n0 + mr;
  #pragma unroll
  for (int r = 0; r < 8; ++r){
    int row0 = m0 + r + half*8;
    int row1 = row0 + 16;
    float b0v = bias ? bias[cn]      : 0.f;
    float b1v = bias ? bias[cn + 16] : 0.f;
    if (row0 < M){
      if (cn      < N) C[(long)row0*ldc + cn]      = acc00[r] + b0v;
      if (cn + 16 < N) C[(long)row0*ldc + cn + 16] = acc01[r] + b1v;
    }
    if (row1 < M){
      if (cn      < N) C[(long)row1*ldc + cn]      = acc10[r] + b0v;
      if (cn + 16 < N) C[(long)row1*ldc + cn + 16] = acc11[r] + b1v;
    }
  }
}

// ---------------- scores = v . tanh(s1+s2), mask, softmax -> a (padded bf16) -
__global__ void attn_softmax_k(const float* __restrict__ s1, const float* __restrict__ s2,
                               const float* __restrict__ vvec,
                               const unsigned char* __restrict__ mask,
                               bf16* __restrict__ abf){
  int bq = blockIdx.x;            // b*LQP + q
  int b  = bq / LQP, q = bq % LQP;
  int tid = threadIdx.x;          // 256 threads
  bf16* rowp = abf + ((long)b*LQP + q)*LKP;
  if (q >= LL){                   // padded query rows: zero
    for (int vp = tid; vp < LKP; vp += 256) rowp[vp] = (bf16)0.f;
    return;
  }
  __shared__ float s2s[HH], vs[HH], sc[LL], red[256];
  if (tid < HH){ s2s[tid] = s2[((long)b*LL + q)*HH + tid]; vs[tid] = vvec[tid]; }
  __syncthreads();
  for (int vp = tid; vp < LL; vp += 256){
    const float* s1p = s1 + ((long)b*LL + vp)*HH;
    float acc = 0.f;
    #pragma unroll 4
    for (int h = 0; h < HH; ++h) acc += vs[h]*tanhf(s1p[h] + s2s[h]);
    sc[vp] = mask[b*LL + vp] ? acc : NEGF;
  }
  __syncthreads();
  float m = NEGF;
  for (int vp = tid; vp < LL; vp += 256) m = fmaxf(m, sc[vp]);
  red[tid] = m; __syncthreads();
  for (int s = 128; s > 0; s >>= 1){
    if (tid < s) red[tid] = fmaxf(red[tid], red[tid+s]);
    __syncthreads();
  }
  float mx = red[0]; __syncthreads();
  float sum = 0.f;
  for (int vp = tid; vp < LL; vp += 256){
    float e = __expf(sc[vp]-mx); sc[vp] = e; sum += e;
  }
  red[tid] = sum; __syncthreads();
  for (int s = 128; s > 0; s >>= 1){
    if (tid < s) red[tid] += red[tid+s];
    __syncthreads();
  }
  float inv = 1.f/red[0];
  for (int vp = tid; vp < LKP; vp += 256)
    rowp[vp] = (vp < LL) ? (bf16)(sc[vp]*inv) : (bf16)0.f;
}

// ---------------- x = sigmoid(gproj)*rnn_in ; also length-reversed copy ------
__global__ void gate_apply_k(const float* __restrict__ rnn, const float* __restrict__ gproj,
                             const int* __restrict__ qlens,
                             bf16* __restrict__ xbf, bf16* __restrict__ xrevbf){
  long i = (long)blockIdx.x*blockDim.x + threadIdx.x;
  if (i >= (long)BL*D2) return;
  int d = (int)(i % D2);
  int row = (int)(i / D2);
  int b = row / LL, t = row % LL;
  xbf[i] = (bf16)(rnn[i]*sigmoidf_(gproj[i]));
  int len = qlens[b];
  int rt = (t < len) ? (len-1-t) : t;            // flip valid prefix
  long j = ((long)(b*LL + rt))*D2 + d;
  xrevbf[i] = (bf16)(rnn[j]*sigmoidf_(gproj[j]));
}

// ---------------- masked GRU, Whh resident in LDS (198KB of 320KB WGP LDS) ---
__global__ void gru_k(const float* __restrict__ xp, const float* __restrict__ Whh,
                      const float* __restrict__ bhh, const int* __restrict__ qlens,
                      float* __restrict__ out, int dir){
  extern __shared__ float lds[];
  float* w  = lds;                 // H3 x 129 (pad -> conflict-free under wave32)
  float* h  = lds + H3*129;        // HH
  float* hp = h + HH;              // H3
  const int tid = threadIdx.x;     // 0..383
  const int b = blockIdx.x;
  for (int i = tid; i < H3*HH; i += H3){
    int r = i / HH, c = i % HH;
    w[r*129 + c] = Whh[i];
  }
  if (tid < HH) h[tid] = 0.f;
  const float bj = bhh[tid];
  const int len = qlens[b];
  __syncthreads();
  for (int t = 0; t < LL; ++t){
    float acc = bj;
    const float* wr = w + tid*129;
    #pragma unroll 8
    for (int k = 0; k < HH; ++k) acc += wr[k]*h[k];
    hp[tid] = acc;
    __syncthreads();
    if (tid < HH){
      long base = ((long)(b*LL + t))*H3;
      float r = sigmoidf_(xp[base + tid]        + hp[tid]);
      float z = sigmoidf_(xp[base + HH + tid]   + hp[HH + tid]);
      float n = tanhf   (xp[base + 2*HH + tid] + r*hp[2*HH + tid]);
      float hv = h[tid];
      float hn = (1.f - z)*n + z*hv;
      float vt = (t < len) ? 1.f : 0.f;
      hv = vt*hn + (1.f - vt)*hv;
      h[tid] = hv;
      int tpos = dir ? ((t < len) ? (len-1-t) : t) : t;
      out[((long)(b*LL + tpos))*(2*HH) + dir*HH + tid] = hv*vt;
    }
    __syncthreads();
  }
}

extern "C" void kernel_launch(void* const* d_in, const int* in_sizes, int n_in,
                              void* d_out, int out_size, void* d_ws, size_t ws_size,
                              hipStream_t stream) {
  const float* uq    = (const float*)d_in[0];
  const float* uvv   = (const float*)d_in[1];
  const unsigned char* mask = (const unsigned char*)d_in[2];
  const int*   qlens = (const int*)d_in[3];
  const float* Wv    = (const float*)d_in[4];
  const float* Wq    = (const float*)d_in[5];
  const float* vvec  = (const float*)d_in[6];
  const float* Wg    = (const float*)d_in[7];
  const float* Wihf  = (const float*)d_in[8];
  const float* Whhf  = (const float*)d_in[9];
  const float* bihf  = (const float*)d_in[10];
  const float* bhhf  = (const float*)d_in[11];
  const float* Wihb  = (const float*)d_in[12];
  const float* Whhb  = (const float*)d_in[13];
  const float* bihb  = (const float*)d_in[14];
  const float* bhhb  = (const float*)d_in[15];
  float* out = (float*)d_out;

  char* ws = (char*)d_ws;
  size_t off = 0;
  auto take = [&](size_t bytes) -> char* {
    char* p = ws + off;
    off = (off + bytes + 255) & ~(size_t)255;
    return p;
  };
  bf16* uq_bf   = (bf16*) take((size_t)BL*DD*sizeof(bf16));
  bf16* uv_bf   = (bf16*) take((size_t)BL*DD*sizeof(bf16));
  bf16* uvT_bf  = (bf16*) take((size_t)BB*DD*LKP*sizeof(bf16));
  bf16* Wv_bf   = (bf16*) take((size_t)HH*DD*sizeof(bf16));
  bf16* Wq_bf   = (bf16*) take((size_t)HH*DD*sizeof(bf16));
  bf16* Wg_bf   = (bf16*) take((size_t)D2*D2*sizeof(bf16));
  bf16* Wihf_bf = (bf16*) take((size_t)H3*D2*sizeof(bf16));
  bf16* Wihb_bf = (bf16*) take((size_t)H3*D2*sizeof(bf16));
  float* s1     = (float*)take((size_t)BL*HH*sizeof(float));
  float* s2     = (float*)take((size_t)BL*HH*sizeof(float));
  bf16* abf     = (bf16*) take((size_t)BB*LQP*LKP*sizeof(bf16));
  float* rnn    = (float*)take((size_t)BL*D2*sizeof(float));
  bf16* rnn_bf  = (bf16*) take((size_t)BL*D2*sizeof(bf16));
  float* gproj  = (float*)take((size_t)BL*D2*sizeof(float));
  bf16* x_bf    = (bf16*) take((size_t)BL*D2*sizeof(bf16));
  bf16* xrev_bf = (bf16*) take((size_t)BL*D2*sizeof(bf16));
  float* xpf    = (float*)take((size_t)BL*H3*sizeof(float));
  float* xpb    = (float*)take((size_t)BL*H3*sizeof(float));
  (void)ws_size; (void)in_sizes; (void)n_in; (void)out_size;

  const int TPB = 256;
  auto cdiv = [](int a, int b){ return (a + b - 1)/b; };

  // f32 -> bf16 packs
  pack_bf16_k<<<cdiv(BL*DD,TPB),TPB,0,stream>>>(uq,   uq_bf,   BL*DD);
  pack_bf16_k<<<cdiv(BL*DD,TPB),TPB,0,stream>>>(uvv,  uv_bf,   BL*DD);
  pack_bf16_k<<<cdiv(HH*DD,TPB),TPB,0,stream>>>(Wv,   Wv_bf,   HH*DD);
  pack_bf16_k<<<cdiv(HH*DD,TPB),TPB,0,stream>>>(Wq,   Wq_bf,   HH*DD);
  pack_bf16_k<<<cdiv(D2*D2,TPB),TPB,0,stream>>>(Wg,   Wg_bf,   D2*D2);
  pack_bf16_k<<<cdiv(H3*D2,TPB),TPB,0,stream>>>(Wihf, Wihf_bf, H3*D2);
  pack_bf16_k<<<cdiv(H3*D2,TPB),TPB,0,stream>>>(Wihb, Wihb_bf, H3*D2);
  transpose_uv_k<<<cdiv(BB*DD*LKP,TPB),TPB,0,stream>>>(uv_bf, uvT_bf);

  // s1 = u_value @ Wv^T, s2 = u_query @ Wq^T   (M=2400, N=128, K=256)
  gemm_wmma_k<<<dim3(BL/32, HH/32, 1), 32, 0, stream>>>(
      uv_bf, Wv_bf, nullptr, s1, BL, HH, DD, DD, DD, HH, 0, 0, 0);
  gemm_wmma_k<<<dim3(BL/32, HH/32, 1), 32, 0, stream>>>(
      uq_bf, Wq_bf, nullptr, s2, BL, HH, DD, DD, DD, HH, 0, 0, 0);

  // scores + mask + softmax -> a (bf16, padded LQP x LKP per batch)
  attn_softmax_k<<<BB*LQP, 256, 0, stream>>>(s1, s2, vvec, mask, abf);

  // rnn_in = [u_query, c]; c = a @ uvT^T (batched; writes upper 256 cols)
  concat_query_k<<<cdiv(BL*DD,TPB),TPB,0,stream>>>(uq, rnn);
  gemm_wmma_k<<<dim3(LQP/32, DD/32, BB), 32, 0, stream>>>(
      abf, uvT_bf, nullptr, rnn + DD, LL, DD, LKP, LKP, LKP, D2,
      (long)LQP*LKP, (long)DD*LKP, (long)LL*D2);

  // gate projection gproj = rnn_in @ Wg^T
  pack_bf16_k<<<cdiv(BL*D2,TPB),TPB,0,stream>>>(rnn, rnn_bf, BL*D2);
  gemm_wmma_k<<<dim3(BL/32, D2/32, 1), 32, 0, stream>>>(
      rnn_bf, Wg_bf, nullptr, gproj, BL, D2, D2, D2, D2, D2, 0, 0, 0);

  // x = sigmoid(gproj)*rnn_in (bf16) + length-reversed copy for backward GRU
  gate_apply_k<<<cdiv(BL*D2,TPB),TPB,0,stream>>>(rnn, gproj, qlens, x_bf, xrev_bf);

  // GRU input projections xp = x @ Wih^T + bih  (M=2400, N=384, K=512)
  gemm_wmma_k<<<dim3(BL/32, H3/32, 1), 32, 0, stream>>>(
      x_bf,    Wihf_bf, bihf, xpf, BL, H3, D2, D2, D2, H3, 0, 0, 0);
  gemm_wmma_k<<<dim3(BL/32, H3/32, 1), 32, 0, stream>>>(
      xrev_bf, Wihb_bf, bihb, xpb, BL, H3, D2, D2, D2, H3, 0, 0, 0);

  // Recurrences: Whh (3H x H, padded) resident in LDS for all 300 steps
  size_t gruLds = (size_t)(H3*129 + HH + H3)*sizeof(float);
  gru_k<<<BB, H3, gruLds, stream>>>(xpf, Whhf, bhhf, qlens, out, 0);
  gru_k<<<BB, H3, gruLds, stream>>>(xpb, Whhb, bhhb, qlens, out, 1);
}
